// MultiHeadAttention_3315714752788
// MI455X (gfx1250) — matrix operand order repaired and checked by reading, not compiled, not run on hardware
//
#include <hip/hip_runtime.h>

typedef __attribute__((ext_vector_type(16))) __bf16 v16bf;
typedef __attribute__((ext_vector_type(8)))  float  v8f;
typedef __attribute__((ext_vector_type(4)))  float  vf4;   // native vector (NT-store legal)

#define B_  4
#define L_  2048
#define D_  1024
#define H_  16
#define DK_ 64
#define BM  32   // query rows per workgroup

__device__ __forceinline__ float wave_max(float x) {
  #pragma unroll
  for (int off = 16; off >= 1; off >>= 1)
    x = fmaxf(x, __shfl_xor(x, off, 32));
  return x;
}
__device__ __forceinline__ float wave_sum(float x) {
  #pragma unroll
  for (int off = 16; off >= 1; off >>= 1)
    x += __shfl_xor(x, off, 32);
  return x;
}

// One workgroup = (b, h, 32-row query block). 8 waves of 32.
__global__ __launch_bounds__(256, 1) void attn_kernel(
    const float* __restrict__ q, const float* __restrict__ k,
    const float* __restrict__ v, const unsigned char* __restrict__ mask,
    float* __restrict__ o, float* __restrict__ attn) {
  __shared__ float sP[BM][L_];            // 32 x 2048 fp32 = 256 KB (WGP has 320 KB)
  const int b = blockIdx.z, h = blockIdx.y, mblk = blockIdx.x;
  const int wave = threadIdx.x >> 5, lane = threadIdx.x & 31;
  const int l15  = lane & 15;             // row (A) or col (B/C) within tile
  const int hi   = lane >> 4;             // lane-half select per ISA layout

  // ---------------- Phase 1: S = (Q K^T) / sqrt(dk) into LDS ----------------
  {
    const int mt   = wave & 1;            // which 16-row M tile
    const int row0 = mblk * BM + mt * 16;
    // Q A-fragments (16x32 bf16 each): a0 covers K=0..31, a1 covers K=32..63.
    v16bf a0, a1;
    const float* qrow = q + ((size_t)(b * L_ + row0 + l15)) * D_ + h * DK_;
    #pragma unroll
    for (int i = 0; i < 8; ++i) {
      const int kidx = hi * 16 + 2 * i;
      a0[2*i]   = (__bf16)qrow[kidx];
      a0[2*i+1] = (__bf16)qrow[kidx + 1];
      a1[2*i]   = (__bf16)qrow[32 + kidx];
      a1[2*i+1] = (__bf16)qrow[32 + kidx + 1];
    }
    for (int n = (wave >> 1); n < L_ / 16; n += 4) {
      const int col0 = n * 16;
      // K^T B-fragments (32x16): B[kk][ncol] = K[col0+ncol][kk]
      v16bf b0, b1;
      const float* krow = k + ((size_t)(b * L_ + col0 + l15)) * D_ + h * DK_;
      #pragma unroll
      for (int i = 0; i < 8; ++i) {
        const int kk = hi * 16 + 2 * i;
        b0[2*i]   = (__bf16)krow[kk];
        b0[2*i+1] = (__bf16)krow[kk + 1];
        b1[2*i]   = (__bf16)krow[32 + kk];
        b1[2*i+1] = (__bf16)krow[32 + kk + 1];
      }
      v8f c = {};
      c = __builtin_amdgcn_wmma_f32_16x16x32_bf16(false, a0, false, b0,
                                                  (short)0, c, false, false);
      c = __builtin_amdgcn_wmma_f32_16x16x32_bf16(false, a1, false, b1,
                                                  (short)0, c, false, false);
      #pragma unroll
      for (int r = 0; r < 8; ++r)                      // C layout: M = r + 8*hi, N = l15
        sP[mt * 16 + hi * 8 + r][col0 + l15] = c[r] * 0.125f;   // 1/sqrt(64)
    }
  }
  __syncthreads();

  // ------- Phase 2: row softmax + stream attn to HBM (b128, NT stores) -------
  for (int r = wave; r < BM; r += 8) {
    const int grow = mblk * BM + r;
    const uchar4* mr = (const uchar4*)(mask + ((size_t)(b * L_ + grow)) * L_);
    vf4* srow = (vf4*)sP[r];
    float mx = -3.4e38f;
    for (int c4 = lane; c4 < L_ / 4; c4 += 32) {
      vf4 s = srow[c4];
      const uchar4 m4 = mr[c4];
      if (m4.x) s.x = -1e9f;
      if (m4.y) s.y = -1e9f;
      if (m4.z) s.z = -1e9f;
      if (m4.w) s.w = -1e9f;
      srow[c4] = s;
      mx = fmaxf(mx, fmaxf(fmaxf(s.x, s.y), fmaxf(s.z, s.w)));
    }
    mx = wave_max(mx);
    float sum = 0.f;
    for (int c4 = lane; c4 < L_ / 4; c4 += 32) {
      vf4 s = srow[c4];
      s.x = __expf(s.x - mx);
      s.y = __expf(s.y - mx);
      s.z = __expf(s.z - mx);
      s.w = __expf(s.w - mx);
      srow[c4] = s;
      sum += (s.x + s.y) + (s.z + s.w);
    }
    sum = wave_sum(sum);
    const float inv = 1.0f / sum;
    // reference attn layout: [h*B + b][l][m]; non-temporal: 1 GiB stream, keep L2 for K/V
    vf4* arow = (vf4*)(attn + (((size_t)(h * B_ + b)) * L_ + grow) * (size_t)L_);
    for (int c4 = lane; c4 < L_ / 4; c4 += 32) {
      vf4 p = srow[c4] * inv;
      srow[c4] = p;
      __builtin_nontemporal_store(p, arow + c4);   // global_store_b128, TH=NT
    }
  }
  __syncthreads();

  // ---------------- Phase 3: O = P @ V from LDS (no HBM re-read) ----------------
  {
    const int mt = wave & 1, nt = wave >> 1;   // 2x4 grid of 16x16 output tiles
    v8f acc = {};
    for (int kt = 0; kt < L_ / 32; ++kt) {
      v16bf pa, vb;
      const float* prow = &sP[mt * 16 + l15][kt * 32];
      #pragma unroll
      for (int i = 0; i < 8; ++i) {
        const int kidx = hi * 16 + 2 * i;
        pa[2*i]   = (__bf16)prow[kidx];
        pa[2*i+1] = (__bf16)prow[kidx + 1];
      }
      // V B-fragment: B[kk][ncol] = V[kt*32 + kk][h*64 + nt*16 + ncol]
      const float* vcol = v + ((size_t)(b * L_ + kt * 32 + hi * 16)) * D_
                            + h * DK_ + nt * 16 + l15;
      #pragma unroll
      for (int i = 0; i < 8; ++i) {
        vb[2*i]   = (__bf16)vcol[(2 * i) * D_];
        vb[2*i+1] = (__bf16)vcol[(2 * i + 1) * D_];
      }
      acc = __builtin_amdgcn_wmma_f32_16x16x32_bf16(false, pa, false, vb,
                                                    (short)0, acc, false, false);
    }
    #pragma unroll
    for (int r = 0; r < 8; ++r) {
      const int grow = mblk * BM + mt * 16 + hi * 8 + r;
      o[((size_t)(b * L_ + grow)) * D_ + h * DK_ + nt * 16 + l15] = acc[r];
    }
  }
}

// Residual + LayerNorm over D=1024; reads attention output in place from d_out.
__global__ __launch_bounds__(256, 1) void ln_kernel(
    const float* __restrict__ q, const float* __restrict__ gamma,
    const float* __restrict__ beta, float* __restrict__ y) {
  __shared__ float red[2][8];
  const int row = blockIdx.x, tid = threadIdx.x;
  const int wave = tid >> 5, lane = tid & 31;
  vf4* yrow = (vf4*)(y + (size_t)row * D_);
  const vf4* qrow = (const vf4*)(q + (size_t)row * D_);
  const vf4 o4 = yrow[tid];
  const vf4 q4 = qrow[tid];
  vf4 x = o4 + q4;
  float s  = x.x + x.y + x.z + x.w;
  float ss = x.x * x.x + x.y * x.y + x.z * x.z + x.w * x.w;
  s = wave_sum(s);
  ss = wave_sum(ss);
  if (lane == 0) { red[0][wave] = s; red[1][wave] = ss; }
  __syncthreads();
  if (wave == 0) {
    float a  = (lane < 8) ? red[0][lane] : 0.f;
    float bb = (lane < 8) ? red[1][lane] : 0.f;
    a = wave_sum(a);
    bb = wave_sum(bb);
    if (lane == 0) { red[0][0] = a; red[1][0] = bb; }
  }
  __syncthreads();
  const float mean = red[0][0] * (1.0f / D_);
  const float var  = red[1][0] * (1.0f / D_) - mean * mean;
  const float rs   = rsqrtf(var + 1e-5f);
  const vf4 g4 = ((const vf4*)gamma)[tid];
  const vf4 b4 = ((const vf4*)beta)[tid];
  vf4 out = (x - mean) * rs * g4 + b4;
  yrow[tid] = out;
}

extern "C" void kernel_launch(void* const* d_in, const int* in_sizes, int n_in,
                              void* d_out, int out_size, void* d_ws, size_t ws_size,
                              hipStream_t stream) {
  const float* q = (const float*)d_in[0];
  const float* k = (const float*)d_in[1];
  const float* v = (const float*)d_in[2];
  const unsigned char* mask = (const unsigned char*)d_in[3];  // jnp bool = 1 byte
  const float* gamma = (const float*)d_in[4];
  const float* beta  = (const float*)d_in[5];
  float* y    = (float*)d_out;                 // [B, L, D]
  float* attn = y + (size_t)B_ * L_ * D_;      // [H*B, L, L]

  dim3 g1(L_ / BM, H_, B_);
  attn_kernel<<<g1, 256, 0, stream>>>(q, k, v, mask, y, attn);
  ln_kernel<<<B_ * L_, 256, 0, stream>>>(q, gamma, beta, y);
}
